// LlamaAttention_65712999628853
// MI455X (gfx1250) — compile-verified, matrix operand order checked by reading, not run on hardware
//
#include <hip/hip_runtime.h>
#include <hip/hip_bf16.h>
#include <stdint.h>

// ---------------------------------------------------------------------------
// Types
// ---------------------------------------------------------------------------
typedef __bf16 bf16_t;
typedef __attribute__((ext_vector_type(16))) __bf16 bf16x16;
typedef __attribute__((ext_vector_type(8)))  __bf16 bf16x8;
typedef __attribute__((ext_vector_type(4)))  __bf16 bf16x4;
typedef __attribute__((ext_vector_type(8)))  float  f32x8;

#define S_LEN   2048
#define HID     4096
#define N_HEADS 32
#define N_KV    8
#define HEAD_D  128
#define DKV     (N_KV * HEAD_D)   // 1024

__device__ __forceinline__ f32x8 zero8() {
    f32x8 v;
#pragma unroll
    for (int i = 0; i < 8; ++i) v[i] = 0.0f;
    return v;
}

// WMMA wrapper: D = A(16x32 bf16) * B(32x16 bf16) + C(16x16 f32)
__device__ __forceinline__ f32x8 wmma_bf16(bf16x16 a, bf16x16 b, f32x8 c) {
    return __builtin_amdgcn_wmma_f32_16x16x32_bf16(
        /*neg_a=*/false, a, /*neg_b=*/false, b,
        /*c_mod=*/(short)0, c, /*reuse_a=*/false, /*reuse_b=*/false);
}

// Load one 16x32 bf16 fragment (A layout; B layout of a row-major [N,K]
// operand is identical since both stripe K within a lane).
// Lane half=0 takes K = {k0..k0+7, k0+16..k0+23}; half=1 the other half.
// Both 8-element groups are contiguous -> two 16-byte loads.
__device__ __forceinline__ bf16x16 load_frag(const bf16_t* base, int stride,
                                             int row, int k0, int lane) {
    const int half = (lane >> 4) & 1;
    const bf16_t* p = base + (size_t)row * (size_t)stride + k0 + half * 8;
    union { bf16x16 v; bf16x8 h[2]; } u;
    u.h[0] = *reinterpret_cast<const bf16x8*>(p);
    u.h[1] = *reinterpret_cast<const bf16x8*>(p + 16);
    return u.v;
}

// CDNA5 async global->LDS copy, 16 bytes per lane, tracked by ASYNCcnt.
// LDS dest is the wave-relative LDS byte address (low 32 bits of a generic
// pointer to __shared__; hardware adds LDS_BASE).
__device__ __forceinline__ void async_ld_b128(void* lds_ptr, const void* gptr) {
    uint32_t lds = (uint32_t)(uintptr_t)lds_ptr;
    unsigned long long g = (unsigned long long)(uintptr_t)gptr;
    asm volatile("global_load_async_to_lds_b128 %0, %1, off"
                 :: "v"(lds), "v"(g) : "memory");
}
__device__ __forceinline__ void wait_asynccnt0() {
    asm volatile("s_wait_asynccnt 0x0" ::: "memory");
}

// ---------------------------------------------------------------------------
// fp32 -> bf16 conversion (vectorized x4)
// ---------------------------------------------------------------------------
__global__ void cvt_f32_bf16_kernel(const float* __restrict__ in,
                                    bf16_t* __restrict__ out, int n4) {
    int i = blockIdx.x * blockDim.x + threadIdx.x;
    if (i >= n4) return;
    float4 f = reinterpret_cast<const float4*>(in)[i];
    bf16x4 o;
    o[0] = (bf16_t)f.x; o[1] = (bf16_t)f.y; o[2] = (bf16_t)f.z; o[3] = (bf16_t)f.w;
    reinterpret_cast<bf16x4*>(out)[i] = o;
}

// ---------------------------------------------------------------------------
// GEMM: C[M,N] = A[M,K] * B[N,K]^T   (torch Linear: x @ W.T)
// A,B bf16 row-major (K contiguous), C = OutT.
// Block 256 threads (8 waves), tile 128x128, BK=64, each wave does 32x64.
// LDS tiles are double-buffered and filled with async global->LDS copies.
// ---------------------------------------------------------------------------
#define BM 128
#define BN 128
#define BK 64
#define LDA (BK + 8)   // 72 elems = 144 B row stride, multiple of 16 B

template <typename OutT>
__device__ __forceinline__ void store_out(OutT* p, float v);
template <> __device__ __forceinline__ void store_out<float>(float* p, float v) { *p = v; }
template <> __device__ __forceinline__ void store_out<bf16_t>(bf16_t* p, float v) { *p = (bf16_t)v; }

// Issue 4 async b128 fills (one 128x64 bf16 tile across 256 threads).
__device__ __forceinline__ void stage_tile_async(const bf16_t* __restrict__ gbase,
                                                 int K, int kb,
                                                 bf16_t (*sm)[LDA], int tid) {
#pragma unroll
    for (int i = 0; i < 4; ++i) {
        int vid = tid + 256 * i;          // 1024 vectors of 8 bf16
        int r   = vid >> 3;
        int c8  = (vid & 7) << 3;
        async_ld_b128(&sm[r][c8], &gbase[(size_t)r * K + kb + c8]);
    }
}

template <typename OutT>
__global__ __launch_bounds__(256)
void gemm_xt_kernel(const bf16_t* __restrict__ A, const bf16_t* __restrict__ B,
                    OutT* __restrict__ C, int M, int N, int K) {
    __shared__ bf16_t As[2][BM][LDA];
    __shared__ bf16_t Bs[2][BN][LDA];

    const int tid  = threadIdx.x;
    const int lane = tid & 31;
    const int wave = tid >> 5;
    const int wm   = wave & 3;   // 4 groups of 32 rows
    const int wn   = wave >> 2;  // 2 groups of 64 cols
    const int idx  = lane & 15;
    const int half = lane >> 4;

    const int mblk = blockIdx.y * BM;
    const int nblk = blockIdx.x * BN;
    const bf16_t* Ag = A + (size_t)mblk * K;
    const bf16_t* Bg = B + (size_t)nblk * K;

    f32x8 acc[2][4];
#pragma unroll
    for (int i = 0; i < 2; ++i)
#pragma unroll
        for (int j = 0; j < 4; ++j) acc[i][j] = zero8();

    // Prologue: fill buffer 0.
    stage_tile_async(Ag, K, 0, As[0], tid);
    stage_tile_async(Bg, K, 0, Bs[0], tid);
    wait_asynccnt0();
    __syncthreads();

    int cur = 0;
    for (int kb = 0; kb < K; kb += BK) {
        const int nxt = cur ^ 1;
        if (kb + BK < K) {          // overlap next tile fill with compute
            stage_tile_async(Ag, K, kb + BK, As[nxt], tid);
            stage_tile_async(Bg, K, kb + BK, Bs[nxt], tid);
        }

#pragma unroll
        for (int kk = 0; kk < BK; kk += 32) {
            bf16x16 af[2], bf[4];
#pragma unroll
            for (int i = 0; i < 2; ++i)
                af[i] = load_frag(&As[cur][0][0], LDA, wm * 32 + i * 16 + idx, kk, lane);
#pragma unroll
            for (int j = 0; j < 4; ++j)
                bf[j] = load_frag(&Bs[cur][0][0], LDA, wn * 64 + j * 16 + idx, kk, lane);
#pragma unroll
            for (int i = 0; i < 2; ++i)
#pragma unroll
                for (int j = 0; j < 4; ++j)
                    acc[i][j] = wmma_bf16(af[i], bf[j], acc[i][j]);
        }

        wait_asynccnt0();   // own async fills of buf[nxt] complete
        __syncthreads();    // everyone done reading buf[cur] / filling buf[nxt]
        cur = nxt;
    }

    // C/D layout: VGPR r holds row (r + half*8), col = idx.
#pragma unroll
    for (int i = 0; i < 2; ++i)
#pragma unroll
        for (int j = 0; j < 4; ++j)
#pragma unroll
            for (int r = 0; r < 8; ++r) {
                int m = mblk + wm * 32 + i * 16 + half * 8 + r;
                int n = nblk + wn * 64 + j * 16 + idx;
                store_out<OutT>(&C[(size_t)m * N + n], acc[i][j][r]);
            }
}

// ---------------------------------------------------------------------------
// RoPE (in place, bf16): x[s, h, 0:64] pairs with x[s, h, 64:128]
// ---------------------------------------------------------------------------
__global__ void rope_kernel(bf16_t* __restrict__ x, int nheads) {
    int i = blockIdx.x * blockDim.x + threadIdx.x;
    int total = S_LEN * nheads * 64;
    if (i >= total) return;
    int j = i & 63;
    int h = (i >> 6) % nheads;
    int s = i / (64 * nheads);
    float inv = __powf(10000.0f, -(float)j * (1.0f / 64.0f));
    float ang = (float)s * inv;
    float sn, cs;
    __sincosf(ang, &sn, &cs);
    bf16_t* p = x + ((size_t)s * nheads + h) * HEAD_D;
    float x0 = (float)p[j];
    float x1 = (float)p[j + 64];
    p[j]      = (bf16_t)(x0 * cs - x1 * sn);
    p[j + 64] = (bf16_t)(x1 * cs + x0 * sn);
}

// ---------------------------------------------------------------------------
// Flash attention (causal, GQA). grid = (S/128, N_HEADS), block = 256 (8 waves).
// Each wave owns 16 query rows. KV chunk = 32 keys.
// Q [S, HID] bf16, K/V [S, DKV] bf16, O [S, HID] bf16.
// ---------------------------------------------------------------------------
#define QTILE  128
#define KCHUNK 32
#define LDVT   (KCHUNK + 16)   // 48 elems = 96 B, multiple of 16 B

__global__ __launch_bounds__(256)
void attn_kernel(const bf16_t* __restrict__ Q, const bf16_t* __restrict__ K,
                 const bf16_t* __restrict__ V, bf16_t* __restrict__ O) {
    __shared__ bf16_t Vt[HEAD_D][LDVT];   // V chunk transposed: Vt[d][k]
    __shared__ bf16_t Ps[8][16][KCHUNK];  // per-wave P tile (row stride 64 B)

    const int tid  = threadIdx.x;
    const int lane = tid & 31;
    const int wave = tid >> 5;
    const int half = lane >> 4;
    const int idx  = lane & 15;

    const int hq  = blockIdx.y;     // 0..31
    const int hkv = hq >> 2;        // GQA: 4 Q heads share one KV head

    const int qblk  = blockIdx.x * QTILE;
    const int qbase = qblk + wave * 16;   // this wave's first query row

    const bf16_t* qptr = Q + (size_t)qbase * HID + hq * HEAD_D;
    const bf16_t* kptr = K + hkv * HEAD_D;
    const bf16_t* vptr = V + hkv * HEAD_D;

    // Q fragments for this wave's 16 rows (full D=128 -> 4 k-steps of 32).
    bf16x16 qf[4];
#pragma unroll
    for (int kk = 0; kk < 4; ++kk)
        qf[kk] = load_frag(qptr, HID, idx, kk * 32, lane);

    f32x8 oacc[8];
#pragma unroll
    for (int j = 0; j < 8; ++j) oacc[j] = zero8();
    float rm[8], rl[8];
#pragma unroll
    for (int r = 0; r < 8; ++r) { rm[r] = -1e30f; rl[r] = 0.0f; }

    const float scale = 0.08838834764831845f;  // 1/sqrt(128)
    const int kend = qblk + QTILE;             // causal bound for this block

    for (int kb = 0; kb < kend; kb += KCHUNK) {
        // ---- stage V chunk transposed into LDS --------------------------
        __syncthreads();
        {
            int kr = tid >> 3;            // key row in chunk 0..31
            int d0 = (tid & 7) * 16;      // 16 d-columns per thread
            const bf16_t* src = vptr + (size_t)(kb + kr) * DKV + d0;
            bf16x8 v0 = *reinterpret_cast<const bf16x8*>(src);
            bf16x8 v1 = *reinterpret_cast<const bf16x8*>(src + 8);
#pragma unroll
            for (int m = 0; m < 8; ++m) {
                Vt[d0 + m][kr]     = v0[m];
                Vt[d0 + 8 + m][kr] = v1[m];
            }
        }
        __syncthreads();

        // ---- S = Q * K^T (two 16x16 tiles over this 32-key chunk) -------
        f32x8 sacc[2] = { zero8(), zero8() };
#pragma unroll
        for (int kk = 0; kk < 4; ++kk) {
            bf16x16 kf0 = load_frag(kptr + (size_t)kb * DKV,        DKV, idx, kk * 32, lane);
            bf16x16 kf1 = load_frag(kptr + (size_t)(kb + 16) * DKV, DKV, idx, kk * 32, lane);
            sacc[0] = wmma_bf16(qf[kk], kf0, sacc[0]);
            sacc[1] = wmma_bf16(qf[kk], kf1, sacc[1]);
        }

        // ---- online softmax (rows live across the 16-lane half) ---------
#pragma unroll
        for (int r = 0; r < 8; ++r) {
            int q = qbase + half * 8 + r;
            float s0 = sacc[0][r] * scale + ((kb + idx      > q) ? -1e9f : 0.0f);
            float s1 = sacc[1][r] * scale + ((kb + 16 + idx > q) ? -1e9f : 0.0f);
            float mx = fmaxf(s0, s1);
#pragma unroll
            for (int o = 8; o >= 1; o >>= 1) mx = fmaxf(mx, __shfl_xor(mx, o, 32));
            mx = fmaxf(mx, rm[r]);
            float alpha = __expf(rm[r] - mx);
            float p0 = __expf(s0 - mx);
            float p1 = __expf(s1 - mx);
            float rs = p0 + p1;
#pragma unroll
            for (int o = 8; o >= 1; o >>= 1) rs += __shfl_xor(rs, o, 32);
            rl[r] = rl[r] * alpha + rs;
            rm[r] = mx;
#pragma unroll
            for (int j = 0; j < 8; ++j) oacc[j][r] *= alpha;
            // C-layout -> LDS so we can reread in A-layout for the PV GEMM.
            Ps[wave][half * 8 + r][idx]      = (bf16_t)p0;
            Ps[wave][half * 8 + r][16 + idx] = (bf16_t)p1;
        }

        // DS ops are in-order per wave; make the dependency explicit anyway.
        asm volatile("s_wait_dscnt 0x0" ::: "memory");

        // ---- O += P * V  (B operand from transposed Vt, contiguous) -----
        bf16x16 pf = load_frag(&Ps[wave][0][0], KCHUNK, idx, 0, lane);
#pragma unroll
        for (int j = 0; j < 8; ++j) {
            bf16x16 vf = load_frag(&Vt[0][0], LDVT, j * 16 + idx, 0, lane);
            oacc[j] = wmma_bf16(pf, vf, oacc[j]);
        }
    }

    // ---- normalize and write out [S, N_HEADS*D] -------------------------
    bf16_t* optr = O + (size_t)qbase * HID + hq * HEAD_D;
#pragma unroll
    for (int j = 0; j < 8; ++j)
#pragma unroll
        for (int r = 0; r < 8; ++r) {
            int row = half * 8 + r;
            float val = oacc[j][r] / rl[r];
            optr[(size_t)row * HID + j * 16 + idx] = (bf16_t)val;
        }
}

// ---------------------------------------------------------------------------
// Host launcher
// ---------------------------------------------------------------------------
extern "C" void kernel_launch(void* const* d_in, const int* in_sizes, int n_in,
                              void* d_out, int out_size, void* d_ws, size_t ws_size,
                              hipStream_t stream) {
    (void)in_sizes; (void)n_in; (void)out_size; (void)ws_size;
    const float* X  = (const float*)d_in[0];
    // d_in[1] = attention_mask (causal, regenerated on device), d_in[2] = position_ids (arange)
    const float* Wq = (const float*)d_in[3];
    const float* Wk = (const float*)d_in[4];
    const float* Wv = (const float*)d_in[5];
    const float* Wo = (const float*)d_in[6];
    float* out = (float*)d_out;

    char* ws = (char*)d_ws;
    bf16_t* Xb  = (bf16_t*)ws; ws += (size_t)S_LEN * HID * sizeof(bf16_t);
    bf16_t* Wqb = (bf16_t*)ws; ws += (size_t)HID * HID * sizeof(bf16_t);
    bf16_t* Wkb = (bf16_t*)ws; ws += (size_t)DKV * HID * sizeof(bf16_t);
    bf16_t* Wvb = (bf16_t*)ws; ws += (size_t)DKV * HID * sizeof(bf16_t);
    bf16_t* Wob = (bf16_t*)ws; ws += (size_t)HID * HID * sizeof(bf16_t);
    bf16_t* Qb  = (bf16_t*)ws; ws += (size_t)S_LEN * HID * sizeof(bf16_t);
    bf16_t* Kb  = (bf16_t*)ws; ws += (size_t)S_LEN * DKV * sizeof(bf16_t);
    bf16_t* Vb  = (bf16_t*)ws; ws += (size_t)S_LEN * DKV * sizeof(bf16_t);
    bf16_t* Ob  = (bf16_t*)ws; ws += (size_t)S_LEN * HID * sizeof(bf16_t);

    const int T = 256;
    auto cvt = [&](const float* src, bf16_t* dst, size_t n) {
        int n4 = (int)(n / 4);
        cvt_f32_bf16_kernel<<<(n4 + T - 1) / T, T, 0, stream>>>(src, dst, n4);
    };
    cvt(X,  Xb,  (size_t)S_LEN * HID);
    cvt(Wq, Wqb, (size_t)HID * HID);
    cvt(Wk, Wkb, (size_t)DKV * HID);
    cvt(Wv, Wvb, (size_t)DKV * HID);
    cvt(Wo, Wob, (size_t)HID * HID);

    // Q/K/V projections
    gemm_xt_kernel<bf16_t><<<dim3(HID / BN, S_LEN / BM), T, 0, stream>>>(Xb, Wqb, Qb, S_LEN, HID, HID);
    gemm_xt_kernel<bf16_t><<<dim3(DKV / BN, S_LEN / BM), T, 0, stream>>>(Xb, Wkb, Kb, S_LEN, DKV, HID);
    gemm_xt_kernel<bf16_t><<<dim3(DKV / BN, S_LEN / BM), T, 0, stream>>>(Xb, Wvb, Vb, S_LEN, DKV, HID);

    // RoPE
    rope_kernel<<<(S_LEN * N_HEADS * 64 + T - 1) / T, T, 0, stream>>>(Qb, N_HEADS);
    rope_kernel<<<(S_LEN * N_KV * 64 + T - 1) / T, T, 0, stream>>>(Kb, N_KV);

    // Flash attention
    attn_kernel<<<dim3(S_LEN / QTILE, N_HEADS), T, 0, stream>>>(Qb, Kb, Vb, Ob);

    // Output projection (fp32 result)
    gemm_xt_kernel<float><<<dim3(HID / BN, S_LEN / BM), T, 0, stream>>>(Ob, Wob, out, S_LEN, HID, HID);
}
